// DeformConv2dModule_26079041421785
// MI455X (gfx1250) — matrix operand order repaired and checked by reading, not compiled, not run on hardware
//
#include <hip/hip_runtime.h>

// Problem constants (reference: N=2, Cin=256, H=W=80, Cout=256, 3x3, stride=1, pad=1)
#define CIN    256
#define COUT   256
#define HH     80
#define WW     80
#define NBATCH 2
#define KTAPS  9
#define HWSZ   (HH * WW)          // 6400
#define M_TOT  (NBATCH * HWSZ)    // 12800
#define KDIM   (CIN * KTAPS)      // 2304

// GEMM tiling
#define MB  64
#define NB  128
#define KB  32
#define LDSS 48                   // LDS row stride in halves (96B: 16B-aligned, staggered banks)

typedef __attribute__((ext_vector_type(16))) _Float16 v16h;
typedef __attribute__((ext_vector_type(8)))  float    v8f;
typedef __attribute__((ext_vector_type(4)))  unsigned int v4u;
typedef __attribute__((ext_vector_type(8)))  int v8i;
typedef __attribute__((ext_vector_type(4)))  int v4i;

__device__ __forceinline__ unsigned short f2h(float f) {
    union { _Float16 h; unsigned short u; } cv;
    cv.h = (_Float16)f;
    return cv.u;
}

// ---------------------------------------------------------------------------
// TDM: issue a 2D-tile global->LDS DMA (tensor_load_to_lds), ISA ch.8 D# layout.
// Tile: tile_h rows x tile_w f16 elements, global row stride = stride_elems.
// LDS destination gets (pad_amount+1)=8 dwords of padding after every 16 dwords
// (one 64B tile row), producing an LDS row stride of 96B = LDSS halves.
// 6-arg builtin on this toolchain:
//   (uint32x4 g0, int32x8 g1, int32x4 g2, int32x4 g3, int32x8 extra, i32 cpol)
// ---------------------------------------------------------------------------
__device__ __forceinline__ void tdm_load_2d_f16(unsigned lds_off, const void* gaddr,
                                                unsigned tile_w, unsigned tile_h,
                                                unsigned stride_elems) {
    const unsigned long long ga = (unsigned long long)(size_t)gaddr;
    v4u g0;
    g0[0] = 1u;                                            // count=1, user mode
    g0[1] = lds_off;                                       // lds_addr (bytes)
    g0[2] = (unsigned)(ga & 0xFFFFFFFFu);                  // global_addr lo
    g0[3] = (unsigned)((ga >> 32) & 0x01FFFFFFu)           // global_addr hi (bits 56:32)
          | (2u << 30);                                    // type = 2 ("image")
    v8i g1;
    g1[0] = (int)((1u << 16)      // data_size = 1 -> 2 bytes
                | (1u << 20)      // pad_enable
                | (3u << 22)      // pad_interval: every 16 dwords (64B row)
                | (7u << 25));    // pad_amount: 8 dwords (32B) -> 96B LDS stride
    g1[1] = (int)((tile_w & 0xFFFFu) << 16);               // tensor_dim0 = tile_w (lo16)
    g1[2] = (int)(((tile_w >> 16) & 0xFFFFu)               // tensor_dim0 hi16
                | ((tile_h & 0xFFFFu) << 16));             // tensor_dim1 = tile_h (lo16)
    g1[3] = (int)(((tile_h >> 16) & 0xFFFFu)               // tensor_dim1 hi16
                | ((tile_w & 0xFFFFu) << 16));             // tile_dim0 = tile_w
    g1[4] = (int)(tile_h & 0xFFFFu);                       // tile_dim1; tile_dim2 = 0
    g1[5] = (int)stride_elems;                             // tensor_dim0_stride lo32
    g1[6] = 0;                                             // stride hi16; dim1_stride lo = 0
    g1[7] = 0;
    const v4i z4 = {0, 0, 0, 0};                           // groups 2/3: 2D tensor
    const v8i z8 = {0, 0, 0, 0, 0, 0, 0, 0};
    __builtin_amdgcn_tensor_load_to_lds(g0, g1, z4, z4, z8, 0);
}

// ---------------------------------------------------------------------------
// Kernel 1: deformable bilinear sampling (implicit im2col) -> f16 A[M][KDIM]
// ck = c*9 + k matches weight's natural (Cout, Cin, 3, 3) flattening.
// ---------------------------------------------------------------------------
__global__ __launch_bounds__(256)
void dcn_sample_kernel(const float* __restrict__ x,
                       const float* __restrict__ off,
                       unsigned short* __restrict__ Ah) {
    const int m  = blockIdx.x;                       // 0..12799
    const int ck = blockIdx.y * 256 + threadIdx.x;   // 0..2303
    const int n  = m / HWSZ;
    const int hw = m - n * HWSZ;
    const int ho = hw / WW;
    const int wo = hw - ho * WW;
    const int k  = ck % KTAPS;
    const int c  = ck / KTAPS;
    const int ky = k / 3;
    const int kx = k - ky * 3;

    const float dy = off[(size_t)(n * 2 * KTAPS + 2 * k + 0) * HWSZ + hw];
    const float dx = off[(size_t)(n * 2 * KTAPS + 2 * k + 1) * HWSZ + hw];

    const float py = (float)(ho - 1 + ky) + dy;      // stride=1, pad=1
    const float px = (float)(wo - 1 + kx) + dx;
    const float y0f = floorf(py);
    const float x0f = floorf(px);
    const float wy = py - y0f;
    const float wx = px - x0f;
    const int y0 = (int)y0f;
    const int x0 = (int)x0f;

    const float* __restrict__ xb = x + (size_t)(n * CIN + c) * HWSZ;

    float v[4];
    const int ys[2] = { y0, y0 + 1 };
    const int xs[2] = { x0, x0 + 1 };
#pragma unroll
    for (int iy = 0; iy < 2; ++iy) {
#pragma unroll
        for (int ix = 0; ix < 2; ++ix) {
            const int yy = ys[iy];
            const int xx = xs[ix];
            const bool valid = (yy >= 0) & (yy < HH) & (xx >= 0) & (xx < WW);
            const int yc = yy < 0 ? 0 : (yy > HH - 1 ? HH - 1 : yy);
            const int xc = xx < 0 ? 0 : (xx > WW - 1 ? WW - 1 : xx);
            const float g = xb[yc * WW + xc];
            v[iy * 2 + ix] = valid ? g : 0.0f;
        }
    }
    const float s = v[0] * (1.f - wy) * (1.f - wx) + v[1] * (1.f - wy) * wx
                  + v[2] * wy * (1.f - wx)         + v[3] * wy * wx;

    Ah[(size_t)m * KDIM + ck] = f2h(s);
}

// ---------------------------------------------------------------------------
// Kernel 2: weight f32 -> f16, layout [Cout][KDIM] (natural flatten)
// ---------------------------------------------------------------------------
__global__ __launch_bounds__(256)
void dcn_wcvt_kernel(const float* __restrict__ w, unsigned short* __restrict__ wh) {
    const int i = blockIdx.x * 256 + threadIdx.x;
    wh[i] = f2h(w[i]);
}

// ---------------------------------------------------------------------------
// Kernel 3: WMMA GEMM  out[m][o] = relu( A[m][:] . B[o][:] + bias[o] )
// Block: 256 threads = 8 wave32 (2 x 4), block tile 64(M) x 128(N).
// Double-buffered LDS tiles, staged by the Tensor Data Mover: wave 0 issues
// the next K-step's DMA right after the publish barrier, so the TDM runs
// concurrently with this K-step's 32 WMMAs. One barrier per K-step.
// ---------------------------------------------------------------------------
__global__ __launch_bounds__(256)
void dcn_gemm_kernel(const unsigned short* __restrict__ Ah,
                     const unsigned short* __restrict__ Bh,
                     const float* __restrict__ bias,
                     float* __restrict__ out) {
    __shared__ alignas(16) unsigned short sA[2][MB * LDSS];   // 2 x 6 KB
    __shared__ alignas(16) unsigned short sB[2][NB * LDSS];   // 2 x 12 KB

    const int tid  = threadIdx.x;
    const int lane = tid & 31;
    const int wave = tid >> 5;
    const int wm   = wave & 1;     // 0..1  -> M offset
    const int wn   = wave >> 1;    // 0..3  -> N offset
    const int lh   = lane >> 4;    // lane half (0/1)
    const int ll   = lane & 15;

    const int m_blk = blockIdx.x * MB;
    const int n_blk = blockIdx.y * NB;

    // LDS byte offsets (generic addr low 32 bits = LDS offset)
    const unsigned ldsA0 = (unsigned)(size_t)(&sA[0][0]);
    const unsigned ldsA1 = (unsigned)(size_t)(&sA[1][0]);
    const unsigned ldsB0 = (unsigned)(size_t)(&sB[0][0]);
    const unsigned ldsB1 = (unsigned)(size_t)(&sB[1][0]);

    const unsigned short* gA = Ah + (size_t)m_blk * KDIM;
    const unsigned short* gB = Bh + (size_t)n_blk * KDIM;

    v8f acc[2][2] = {};

    union Frag { struct { uint4 lo, hi; } q; v16h v; };

    // Prologue: DMA first K-step's tiles into buffer 0
    if (wave == 0) {
        tdm_load_2d_f16(ldsA0, gA, KB, MB, KDIM);
        tdm_load_2d_f16(ldsB0, gB, KB, NB, KDIM);
    }

    int buf = 0;
    for (int kk = 0; kk < KDIM; kk += KB) {
        if (wave == 0)
            __builtin_amdgcn_s_wait_tensorcnt(0);   // tile for this K-step complete
        __syncthreads();                            // publish buf; reclaim buf^1

        // Issue next K-step's DMA into the other buffer; overlaps this
        // step's WMMAs (all readers of buf^1 passed the barrier above).
        if (wave == 0 && kk + KB < KDIM) {
            tdm_load_2d_f16(buf ? ldsA0 : ldsA1, gA + kk + KB, KB, MB, KDIM);
            tdm_load_2d_f16(buf ? ldsB0 : ldsB1, gB + kk + KB, KB, NB, KDIM);
        }

        const unsigned short* __restrict__ cA = &sA[buf][0];
        const unsigned short* __restrict__ cB = &sB[buf][0];

        // A fragments: 16-bit A layout (ISA 7.12.2): lane ll = M row;
        // elements 0..7 = K[8*lh .. 8*lh+7], elements 8..15 = K[16+8*lh ..]
        Frag a[2], b[2];
#pragma unroll
        for (int ti = 0; ti < 2; ++ti) {
            const int row = wm * 32 + ti * 16 + ll;
            a[ti].q.lo = *reinterpret_cast<const uint4*>(&cA[row * LDSS + 8 * lh]);
            a[ti].q.hi = *reinterpret_cast<const uint4*>(&cA[row * LDSS + 16 + 8 * lh]);
        }
        // B fragments: lane ll = N col; elements 0..7 = K[8*lh .. 8*lh+7], elements 8..15 = K[16+8*lh ..]
#pragma unroll
        for (int tj = 0; tj < 2; ++tj) {
            const int col = wn * 32 + tj * 16 + ll;
            b[tj].q.lo = *reinterpret_cast<const uint4*>(&cB[col * LDSS + 8 * lh]);
            b[tj].q.hi = *reinterpret_cast<const uint4*>(&cB[col * LDSS + 16 + 8 * lh]);
        }

#pragma unroll
        for (int ti = 0; ti < 2; ++ti)
#pragma unroll
            for (int tj = 0; tj < 2; ++tj)
                acc[ti][tj] = __builtin_amdgcn_wmma_f32_16x16x32_f16(
                    false, a[ti].v, false, b[tj].v,
                    (short)0, acc[ti][tj], false, false);

        buf ^= 1;
    }

    // Epilogue: C/D layout — VGPR r: lanes 0-15 hold M=r, lanes 16-31 hold M=r+8
#pragma unroll
    for (int ti = 0; ti < 2; ++ti) {
#pragma unroll
        for (int tj = 0; tj < 2; ++tj) {
            const int o  = n_blk + wn * 32 + tj * 16 + ll;
            const float bv = bias[o];
#pragma unroll
            for (int r = 0; r < 8; ++r) {
                const int m  = m_blk + wm * 32 + ti * 16 + lh * 8 + r;
                const int n  = m / HWSZ;
                const int hw = m - n * HWSZ;
                float val = acc[ti][tj][r] + bv;
                val = val > 0.f ? val : 0.f;
                out[(size_t)(n * COUT + o) * HWSZ + hw] = val;
            }
        }
    }
}

// ---------------------------------------------------------------------------
extern "C" void kernel_launch(void* const* d_in, const int* in_sizes, int n_in,
                              void* d_out, int out_size, void* d_ws, size_t ws_size,
                              hipStream_t stream) {
    const float* x    = (const float*)d_in[0];
    const float* off  = (const float*)d_in[1];
    const float* w    = (const float*)d_in[2];
    const float* bias = (const float*)d_in[3];
    float* out = (float*)d_out;

    // workspace: A_f16[12800][2304] (59 MB) then W_f16[256][2304] (1.2 MB)
    unsigned short* Ah = (unsigned short*)d_ws;
    unsigned short* Wh = Ah + (size_t)M_TOT * KDIM;

    dcn_sample_kernel<<<dim3(M_TOT, KDIM / 256), 256, 0, stream>>>(x, off, Ah);
    dcn_wcvt_kernel<<<(COUT * KDIM) / 256, 256, 0, stream>>>(w, Wh);
    dcn_gemm_kernel<<<dim3(M_TOT / MB, COUT / NB), 256, 0, stream>>>(Ah, Wh, bias, out);
}
